// PreLNBertAttention_36807869726806
// MI455X (gfx1250) — matrix operand order repaired and checked by reading, not compiled, not run on hardware
//
#include <hip/hip_runtime.h>

typedef _Float16 h16_t;
typedef __attribute__((ext_vector_type(16))) _Float16 v16h;
typedef __attribute__((ext_vector_type(8)))  float    v8f;

constexpr int SDIM = 2048;   // sequence length
constexpr int HDIM = 512;    // hidden
constexpr int NHEAD = 8;
constexpr int DHEAD = 64;
constexpr int BSROWS = 4 * 2048;  // B*S = 8192

// ---------------------------------------------------------------------------
// WMMA fragment helpers (layouts per CDNA5 ISA 7.12.2, wave32)
// ---------------------------------------------------------------------------

// A fragment: 16x32 f16, lane L holds row m0 + (L%16).
// halves 0..7  = K [k0 + (L/16)*8      .. +7]
// halves 8..15 = K [k0 + (L/16)*8 + 16 .. +7]
__device__ inline v16h load_a16(const h16_t* __restrict__ A, int lda, int m0,
                                int k0, int lane) {
  int row = m0 + (lane & 15);
  int kh  = (lane >> 4) * 8;
  const h16_t* p = A + (size_t)row * lda + k0 + kh;
  union { v16h v; uint4 u[2]; } f;
  f.u[0] = *(const uint4*)(p);
  f.u[1] = *(const uint4*)(p + 16);
  return f.v;
}

// B fragment: 32x16 (KxN) f16 where memory holds Bm[N][K] row-major.
// lane L holds col n0 + (L%16); halves 0..7 = K [k0 + (L/16)*8 .. +7],
// halves 8..15 = K [k0 + (L/16)*8 + 16 .. +7].
__device__ inline v16h load_b16(const h16_t* __restrict__ Bm, int ldb, int n0,
                                int k0, int lane) {
  int n  = n0 + (lane & 15);
  int kb = k0 + ((lane >> 4) << 3);
  const h16_t* p = Bm + (size_t)n * ldb + kb;
  union { v16h v; uint4 u[2]; } f;
  f.u[0] = *(const uint4*)(p);
  f.u[1] = *(const uint4*)(p + 16);
  return f.v;
}

__device__ inline v8f wmma_f16(v16h a, v16h b, v8f c) {
  return __builtin_amdgcn_wmma_f32_16x16x32_f16(false, a, false, b, (short)0,
                                                c, false, false);
}

// ---------------------------------------------------------------------------
// Kernel: f32 -> f16 conversion (weights)
// ---------------------------------------------------------------------------
__global__ void cvt_f32_to_f16(const float* __restrict__ src,
                               h16_t* __restrict__ dst, int n) {
  for (int i = blockIdx.x * blockDim.x + threadIdx.x; i < n;
       i += gridDim.x * blockDim.x)
    dst[i] = (h16_t)src[i];
}

// ---------------------------------------------------------------------------
// Kernel: LayerNorm over H=512. One block (256 thr) per row; each thread
// owns elements t and t+256. Optionally emits an f16 copy for WMMA input.
// ---------------------------------------------------------------------------
__global__ __launch_bounds__(256) void ln_kernel(
    const float* __restrict__ x, const float* __restrict__ g,
    const float* __restrict__ bta, float* __restrict__ out32,
    h16_t* __restrict__ out16) {
  __shared__ float red[256];
  const int row = blockIdx.x;
  const int t   = threadIdx.x;
  const float* xr = x + (size_t)row * HDIM;
  float a0 = xr[t], a1 = xr[t + 256];

  red[t] = a0 + a1;
  __syncthreads();
#pragma unroll
  for (int s = 128; s > 0; s >>= 1) {
    if (t < s) red[t] += red[t + s];
    __syncthreads();
  }
  float mean = red[0] * (1.0f / HDIM);
  __syncthreads();

  float d0 = a0 - mean, d1 = a1 - mean;
  red[t] = d0 * d0 + d1 * d1;
  __syncthreads();
#pragma unroll
  for (int s = 128; s > 0; s >>= 1) {
    if (t < s) red[t] += red[t + s];
    __syncthreads();
  }
  float var = red[0] * (1.0f / HDIM);
  float rs  = rsqrtf(var + 1e-12f);

  float y0 = d0 * rs * g[t] + bta[t];
  float y1 = d1 * rs * g[t + 256] + bta[t + 256];
  size_t base = (size_t)row * HDIM;
  if (out32) { out32[base + t] = y0; out32[base + t + 256] = y1; }
  if (out16) { out16[base + t] = (h16_t)y0; out16[base + t + 256] = (h16_t)y1; }
}

// ---------------------------------------------------------------------------
// Kernel: fused QKV projection GEMM.  out[m,n] = sum_k h16[m,k]*w16[n,k] + b[n]
// blockIdx.z: 0=Q, 1=K (stored [B,NH,S,DH]), 2=V (stored transposed [B,NH,DH,S])
// 256 threads = 8 waves; each wave computes a 16(M) x 64(N) tile.
// ---------------------------------------------------------------------------
__global__ __launch_bounds__(256) void qkv_gemm(
    const h16_t* __restrict__ h16, const h16_t* __restrict__ w16q,
    const h16_t* __restrict__ w16k, const h16_t* __restrict__ w16v,
    const float* __restrict__ bq, const float* __restrict__ bk,
    const float* __restrict__ bv, h16_t* __restrict__ qOut,
    h16_t* __restrict__ kOut, h16_t* __restrict__ vTOut) {
  const int lane = threadIdx.x & 31;
  const int wid  = threadIdx.x >> 5;
  const int m0   = (blockIdx.x * 8 + wid) * 16;
  const int n0   = blockIdx.y * 64;
  const int z    = blockIdx.z;

  const h16_t* W = (z == 0) ? w16q : (z == 1) ? w16k : w16v;
  const float* Bv = (z == 0) ? bq : (z == 1) ? bk : bv;

  v8f acc[4] = {};
  const h16_t* arow = h16 + (size_t)(m0 + (lane & 15)) * HDIM;
  for (int k0 = 0; k0 < HDIM; k0 += 32) {
    if (k0 + 32 < HDIM) __builtin_prefetch(arow + k0 + 32, 0, 0);
    v16h a = load_a16(h16, HDIM, m0, k0, lane);
#pragma unroll
    for (int t = 0; t < 4; ++t) {
      v16h b = load_b16(W, HDIM, n0 + t * 16, k0, lane);
      acc[t] = wmma_f16(a, b, acc[t]);
    }
  }

  const int nl = lane & 15;
  const int hf = lane >> 4;
#pragma unroll
  for (int t = 0; t < 4; ++t) {
    const int n  = n0 + t * 16 + nl;
    const int hh = n >> 6;        // head index
    const int d  = n & 63;        // index within head
    const float bias = Bv[n];
#pragma unroll
    for (int r = 0; r < 8; ++r) {
      const int m  = m0 + r + 8 * hf;        // global row in [0, B*S)
      const int bb = m >> 11;                // batch
      const int s  = m & (SDIM - 1);         // seq position
      const h16_t val = (h16_t)(acc[t][r] + bias);
      if (z == 2) {
        vTOut[(((size_t)bb * NHEAD + hh) * DHEAD + d) * SDIM + s] = val;
      } else {
        h16_t* o = (z == 0) ? qOut : kOut;
        o[(((size_t)bb * NHEAD + hh) * SDIM + s) * DHEAD + d] = val;
      }
    }
  }
}

// ---------------------------------------------------------------------------
// Kernel: flash attention. grid = (S/128, B*NH), block = 256 (8 waves).
// Each wave owns 16 query rows; iterates key chunks of 64 with online softmax.
// Probs staged per-wave in LDS (f16) and reloaded as WMMA A fragments.
// ---------------------------------------------------------------------------
__global__ __launch_bounds__(256) void flash_attn(
    const h16_t* __restrict__ q16, const h16_t* __restrict__ k16,
    const h16_t* __restrict__ vT16, const float* __restrict__ mask,
    h16_t* __restrict__ ctx16) {
  __shared__ h16_t ldsP[8 * 16 * 64];  // 16 KB: per-wave 16x64 probs tile

  const int lane = threadIdx.x & 31;
  const int wid  = threadIdx.x >> 5;
  const int bh   = blockIdx.y;
  const int bb   = bh >> 3;            // batch
  const int hh   = bh & 7;             // head
  const int m0   = blockIdx.x * 128 + wid * 16;

  const h16_t* Q  = q16 + (size_t)bh * SDIM * DHEAD;
  const h16_t* K  = k16 + (size_t)bh * SDIM * DHEAD;
  const h16_t* VT = vT16 + (size_t)bh * DHEAD * SDIM;
  const float* mrow = mask + (size_t)bb * SDIM;
  h16_t* myP = ldsP + wid * (16 * 64);

  // Q fragments for both K-steps (d = 0..31, 32..63); resident all loop.
  v16h qf0 = load_a16(Q, DHEAD, m0, 0, lane);
  v16h qf1 = load_a16(Q, DHEAD, m0, 32, lane);

  v8f   accO[4] = {};
  float mi[8], li[8];
#pragma unroll
  for (int r = 0; r < 8; ++r) { mi[r] = -3.0e38f; li[r] = 0.0f; }

  const int nl = lane & 15;
  const int hf = lane >> 4;

  for (int n0 = 0; n0 < SDIM; n0 += 64) {
    // ---- scores tile: 16 x 64, S = Q K^T * 0.125 + mask ----
    v8f sc[4] = {};
#pragma unroll
    for (int t = 0; t < 4; ++t) {
      v16h b0 = load_b16(K, DHEAD, n0 + t * 16, 0, lane);
      sc[t] = wmma_f16(qf0, b0, sc[t]);
      v16h b1 = load_b16(K, DHEAD, n0 + t * 16, 32, lane);
      sc[t] = wmma_f16(qf1, b1, sc[t]);
    }
    float mk[4];
#pragma unroll
    for (int t = 0; t < 4; ++t) mk[t] = mrow[n0 + t * 16 + nl];
#pragma unroll
    for (int t = 0; t < 4; ++t)
#pragma unroll
      for (int r = 0; r < 8; ++r) sc[t][r] = sc[t][r] * 0.125f + mk[t];

    // ---- online softmax per row (rows live across 16-lane half-waves) ----
#pragma unroll
    for (int r = 0; r < 8; ++r) {
      float vmax = sc[0][r];
#pragma unroll
      for (int t = 1; t < 4; ++t) vmax = fmaxf(vmax, sc[t][r]);
#pragma unroll
      for (int off = 1; off < 16; off <<= 1)
        vmax = fmaxf(vmax, __shfl_xor(vmax, off, 32));
      const float mnew = fmaxf(mi[r], vmax);
      const float corr = __expf(mi[r] - mnew);
      float psum = 0.0f;
#pragma unroll
      for (int t = 0; t < 4; ++t) {
        float p = __expf(sc[t][r] - mnew);
        sc[t][r] = p;
        psum += p;
      }
#pragma unroll
      for (int off = 1; off < 16; off <<= 1) psum += __shfl_xor(psum, off, 32);
      li[r] = li[r] * corr + psum;
      mi[r] = mnew;
#pragma unroll
      for (int t = 0; t < 4; ++t) accO[t][r] *= corr;
    }

    // ---- stage probs (f16) in LDS, row-major 16x64, per-wave region ----
#pragma unroll
    for (int t = 0; t < 4; ++t)
#pragma unroll
      for (int r = 0; r < 8; ++r)
        myP[(r + 8 * hf) * 64 + t * 16 + nl] = (h16_t)sc[t][r];
    asm volatile("s_wait_dscnt 0" ::: "memory");

    // ---- accO += P (16x64) @ V (64x64-chunk over d) ----
#pragma unroll
    for (int ks = 0; ks < 64; ks += 32) {
      union { v16h v; uint4 u[2]; } af;
      const h16_t* pp = myP + (lane & 15) * 64 + ks + (lane >> 4) * 8;
      af.u[0] = *(const uint4*)(pp);
      af.u[1] = *(const uint4*)(pp + 16);
#pragma unroll
      for (int t = 0; t < 4; ++t) {
        v16h b = load_b16(VT, SDIM, t * 16, n0 + ks, lane);
        accO[t] = wmma_f16(af.v, b, accO[t]);
      }
    }
  }

  // ---- normalize and emit ctx (f16, [B,S,H] row-major) ----
  float inv[8];
#pragma unroll
  for (int r = 0; r < 8; ++r) inv[r] = 1.0f / li[r];
#pragma unroll
  for (int t = 0; t < 4; ++t) {
    const int d = t * 16 + nl;
#pragma unroll
    for (int r = 0; r < 8; ++r) {
      const int s = m0 + r + 8 * hf;
      ctx16[((size_t)bb * SDIM + s) * HDIM + hh * DHEAD + d] =
          (h16_t)(accO[t][r] * inv[r]);
    }
  }
}

// ---------------------------------------------------------------------------
// Kernel: output projection + bias + residual(h). y = ctx @ wo^T + bo + h
// ---------------------------------------------------------------------------
__global__ __launch_bounds__(256) void oproj_gemm(
    const h16_t* __restrict__ ctx16, const h16_t* __restrict__ w16o,
    const float* __restrict__ bo, const float* __restrict__ resid,
    float* __restrict__ y) {
  const int lane = threadIdx.x & 31;
  const int wid  = threadIdx.x >> 5;
  const int m0   = (blockIdx.x * 8 + wid) * 16;
  const int n0   = blockIdx.y * 64;

  v8f acc[4] = {};
  const h16_t* arow = ctx16 + (size_t)(m0 + (lane & 15)) * HDIM;
  for (int k0 = 0; k0 < HDIM; k0 += 32) {
    if (k0 + 32 < HDIM) __builtin_prefetch(arow + k0 + 32, 0, 0);
    v16h a = load_a16(ctx16, HDIM, m0, k0, lane);
#pragma unroll
    for (int t = 0; t < 4; ++t) {
      v16h b = load_b16(w16o, HDIM, n0 + t * 16, k0, lane);
      acc[t] = wmma_f16(a, b, acc[t]);
    }
  }

  const int nl = lane & 15;
  const int hf = lane >> 4;
#pragma unroll
  for (int t = 0; t < 4; ++t) {
    const int n = n0 + t * 16 + nl;
    const float bias = bo[n];
#pragma unroll
    for (int r = 0; r < 8; ++r) {
      const int m = m0 + r + 8 * hf;
      const size_t idx = (size_t)m * HDIM + n;
      y[idx] = acc[t][r] + bias + resid[idx];
    }
  }
}

// ---------------------------------------------------------------------------
// Host launcher
// ---------------------------------------------------------------------------
extern "C" void kernel_launch(void* const* d_in, const int* in_sizes, int n_in,
                              void* d_out, int out_size, void* d_ws,
                              size_t ws_size, hipStream_t stream) {
  const float* hidden = (const float*)d_in[0];
  const float* amask  = (const float*)d_in[1];
  const float* ln1_g  = (const float*)d_in[2];
  const float* ln1_b  = (const float*)d_in[3];
  const float* wq     = (const float*)d_in[4];
  const float* bq     = (const float*)d_in[5];
  const float* wk     = (const float*)d_in[6];
  const float* bk     = (const float*)d_in[7];
  const float* wv     = (const float*)d_in[8];
  const float* bv     = (const float*)d_in[9];
  const float* wo     = (const float*)d_in[10];
  const float* bo     = (const float*)d_in[11];
  const float* ln2_g  = (const float*)d_in[12];
  const float* ln2_b  = (const float*)d_in[13];

  char* ws = (char*)d_ws;
  size_t off = 0;
  float*  h32   = (float*)(ws + off);  off += (size_t)BSROWS * HDIM * 4;  // 16 MB
  h16_t*  h16   = (h16_t*)(ws + off);  off += (size_t)BSROWS * HDIM * 2;  // 8 MB
  h16_t*  q16   = (h16_t*)(ws + off);  off += (size_t)BSROWS * HDIM * 2;
  h16_t*  k16   = (h16_t*)(ws + off);  off += (size_t)BSROWS * HDIM * 2;
  h16_t*  vT16  = (h16_t*)(ws + off);  off += (size_t)BSROWS * HDIM * 2;
  h16_t*  ctx16 = (h16_t*)(ws + off);  off += (size_t)BSROWS * HDIM * 2;
  float*  y32   = (float*)(ws + off);  off += (size_t)BSROWS * HDIM * 4;
  h16_t*  w16q  = (h16_t*)(ws + off);  off += (size_t)HDIM * HDIM * 2;
  h16_t*  w16k  = (h16_t*)(ws + off);  off += (size_t)HDIM * HDIM * 2;
  h16_t*  w16v  = (h16_t*)(ws + off);  off += (size_t)HDIM * HDIM * 2;
  h16_t*  w16o  = (h16_t*)(ws + off);  off += (size_t)HDIM * HDIM * 2;

  const int wn = HDIM * HDIM;
  cvt_f32_to_f16<<<dim3(512), 256, 0, stream>>>(wq, w16q, wn);
  cvt_f32_to_f16<<<dim3(512), 256, 0, stream>>>(wk, w16k, wn);
  cvt_f32_to_f16<<<dim3(512), 256, 0, stream>>>(wv, w16v, wn);
  cvt_f32_to_f16<<<dim3(512), 256, 0, stream>>>(wo, w16o, wn);

  ln_kernel<<<dim3(BSROWS), 256, 0, stream>>>(hidden, ln1_g, ln1_b, h32, h16);

  qkv_gemm<<<dim3(BSROWS / 128, HDIM / 64, 3), 256, 0, stream>>>(
      h16, w16q, w16k, w16v, bq, bk, bv, q16, k16, vT16);

  flash_attn<<<dim3(SDIM / 128, 4 * NHEAD), 256, 0, stream>>>(q16, k16, vT16,
                                                              amask, ctx16);

  oproj_gemm<<<dim3(BSROWS / 128, HDIM / 64), 256, 0, stream>>>(ctx16, w16o, bo,
                                                                h32, y32);

  ln_kernel<<<dim3(BSROWS), 256, 0, stream>>>(y32, ln2_g, ln2_b, (float*)d_out,
                                              nullptr);
}
